// SparseMoE_50775103373473
// MI455X (gfx1250) — matrix operand order repaired and checked by reading, not compiled, not run on hardware
//
#include <hip/hip_runtime.h>
#include <math.h>

// Problem dims (fixed by the reference's setup_inputs)
#define TOKENS 8192      // B*S
#define DIM    1024      // D
#define FF     4096      // F
#define NEXP   8         // E
#define CAP    2560      // int(max(1, T*2/8)*1.25), multiple of 128
#define KB     32        // K per WMMA step (f16 16x16x32)

#define MT1 (CAP / 128)  // 20 row tiles
#define NT1 (FF  / 128)  // 32 col tiles (FFN1)
#define NT2 (DIM / 128)  // 8  col tiles (FFN2)
#define LDS_STRIDE 40    // 32 halves + pad (80B rows, 16B aligned, conflict-breaking)

typedef __attribute__((ext_vector_type(16))) _Float16 v16h;
typedef __attribute__((ext_vector_type(8)))  _Float16 v8h;
typedef __attribute__((ext_vector_type(8)))  float    v8f;

union HFrag { v16h v; v8h h[2]; };

__device__ __forceinline__ float gelu_f(float v) {
    const float c = 0.7978845608028654f; // sqrt(2/pi)
    return 0.5f * v * (1.0f + tanhf(c * (v + 0.044715f * v * v * v)));
}

// ---------------------------------------------------------------- zero output
__global__ void zero_kernel(float4* __restrict__ out, int n4) {
    int i = blockIdx.x * blockDim.x + threadIdx.x;
    if (i < n4) out[i] = make_float4(0.f, 0.f, 0.f, 0.f);
}

// ---------------------------------------------------------------- router + top2
// one wave (32 lanes) per token: 8 dot products of length 1024, butterfly reduce
__global__ __launch_bounds__(256) void router_kernel(
        const float* __restrict__ x, const float* __restrict__ wr,
        int* __restrict__ tk_idx, float* __restrict__ tk_gate) {
    int lane = threadIdx.x & 31;
    int wid  = threadIdx.x >> 5;
    int token = blockIdx.x * 8 + wid;
    if (token >= TOKENS) return;
    const float* xr = x + (size_t)token * DIM;
    float acc[NEXP];
#pragma unroll
    for (int e = 0; e < NEXP; ++e) acc[e] = 0.f;
    for (int d = lane; d < DIM; d += 32) {
        float xv = xr[d];
#pragma unroll
        for (int e = 0; e < NEXP; ++e) acc[e] += xv * wr[e * DIM + d];
    }
#pragma unroll
    for (int e = 0; e < NEXP; ++e) {
#pragma unroll
        for (int o = 16; o > 0; o >>= 1) acc[e] += __shfl_xor(acc[e], o, 32);
    }
    if (lane == 0) {
        float v0 = -INFINITY, v1 = -INFINITY; int i0 = 0, i1 = 0;
#pragma unroll
        for (int e = 0; e < NEXP; ++e) {
            float v = acc[e];
            if (v > v0)      { v1 = v0; i1 = i0; v0 = v; i0 = e; }
            else if (v > v1) { v1 = v;  i1 = e; }
        }
        // softmax over the two kept logits (others are -inf)
        float g1 = 1.f / (1.f + __expf(v0 - v1));
        float g0 = 1.f - g1;
        tk_idx[token * 2 + 0] = i0;  tk_idx[token * 2 + 1] = i1;
        tk_gate[token * 2 + 0] = g0; tk_gate[token * 2 + 1] = g1;
    }
}

// ---------------------------------------------------------------- dispatch build
// one wave per expert; ordered prefix over tokens via ballot (matches cumsum+cap)
__global__ void dispatch_kernel(const int* __restrict__ tk_idx, int* __restrict__ disp) {
    int lane = threadIdx.x & 31;
    int e    = threadIdx.x >> 5;   // 8 waves = 8 experts
    int count = 0;
    for (int base = 0; base < TOKENS; base += 32) {
        int t = base + lane;
        bool pred = (tk_idx[t * 2] == e) || (tk_idx[t * 2 + 1] == e);
        unsigned long long m = __ballot(pred);
        int prefix = __popcll(m & ((1ull << lane) - 1ull));
        if (pred) {
            int pos = count + prefix;
            if (pos < CAP) disp[e * CAP + pos] = t;
        }
        count += __popcll(m);
    }
    int start = count < CAP ? count : CAP;
    for (int pos = start + lane; pos < CAP; pos += 32)
        disp[e * CAP + pos] = TOKENS;  // sentinel = padded row
}

// ---------------------------------------------------------------- FFN1: H = gelu(Xg*W1+b1)
__global__ __launch_bounds__(256) void ffn1_kernel(
        const float* __restrict__ x, const float* __restrict__ w1,
        const float* __restrict__ b1, const int* __restrict__ disp,
        _Float16* __restrict__ H) {
    __shared__ _Float16 As[128 * LDS_STRIDE];
    __shared__ _Float16 Bs[128 * LDS_STRIDE];
    __shared__ int toks[128];

    int bid = blockIdx.x;
    int e   = bid / (MT1 * NT1);
    int rem = bid % (MT1 * NT1);
    int m0  = (rem / NT1) * 128;
    int n0  = (rem % NT1) * 128;
    int tid = threadIdx.x;
    int lane = tid & 31, wid = tid >> 5;
    int wm = wid & 3, wn = wid >> 2;          // waves 4(M) x 2(N)

    if (tid < 128) toks[tid] = disp[e * CAP + m0 + tid];
    __syncthreads();

    v8f acc[2][4];
#pragma unroll
    for (int i = 0; i < 2; ++i)
#pragma unroll
        for (int j = 0; j < 4; ++j)
#pragma unroll
            for (int r = 0; r < 8; ++r) acc[i][j][r] = 0.f;

    const float* w1e = w1 + (size_t)e * DIM * FF;
    int arow  = tid >> 1;                // 0..127
    int acol  = (tid & 1) * 16;          // 0 or 16 (k within tile)
    int brow  = tid & 31;                // k within tile
    int bcol0 = (tid >> 5) * 16;         // n chunk

    for (int k0 = 0; k0 < DIM; k0 += KB) {
        // stage A: gathered token rows, fp32 -> f16
        {
            int token = toks[arow];
            _Float16* dst = &As[arow * LDS_STRIDE + acol];
            if (token < TOKENS) {
                const float4* src = (const float4*)(x + (size_t)token * DIM + k0 + acol);
#pragma unroll
                for (int j = 0; j < 4; ++j) {
                    float4 f = src[j];
                    dst[j * 4 + 0] = (_Float16)f.x; dst[j * 4 + 1] = (_Float16)f.y;
                    dst[j * 4 + 2] = (_Float16)f.z; dst[j * 4 + 3] = (_Float16)f.w;
                }
            } else {
#pragma unroll
                for (int j = 0; j < 16; ++j) dst[j] = (_Float16)0.f;
            }
        }
        // stage B: w1 tile [k][n] -> LDS column-major [n][k], fp32 -> f16
        {
            const float* gsrc = w1e + (size_t)(k0 + brow) * FF + n0 + bcol0;
            if (k0 + KB < DIM) __builtin_prefetch(gsrc + (size_t)KB * FF, 0, 1);
            const float4* src = (const float4*)gsrc;
#pragma unroll
            for (int j = 0; j < 4; ++j) {
                float4 f = src[j];
                int n = bcol0 + j * 4;
                Bs[(n + 0) * LDS_STRIDE + brow] = (_Float16)f.x;
                Bs[(n + 1) * LDS_STRIDE + brow] = (_Float16)f.y;
                Bs[(n + 2) * LDS_STRIDE + brow] = (_Float16)f.z;
                Bs[(n + 3) * LDS_STRIDE + brow] = (_Float16)f.w;
            }
        }
        __syncthreads();
        // build fragments: A and B both in contiguous k order (k = 16*half + s)
        HFrag a[2], b[4];
        int sel = lane >> 4;             // half-wave select
#pragma unroll
        for (int i = 0; i < 2; ++i) {
            const v8h* p = (const v8h*)&As[(wm * 32 + i * 16 + (lane & 15)) * LDS_STRIDE];
            a[i].h[0] = p[sel * 2];      // K 0-7 / 16-23
            a[i].h[1] = p[sel * 2 + 1];  // K 8-15 / 24-31
        }
#pragma unroll
        for (int j = 0; j < 4; ++j) {
            const v8h* p = (const v8h*)&Bs[(wn * 64 + j * 16 + (lane & 15)) * LDS_STRIDE + sel * 16];
            b[j].h[0] = p[0];            // K 0-15 / 16-31 contiguous
            b[j].h[1] = p[1];
        }
#pragma unroll
        for (int i = 0; i < 2; ++i)
#pragma unroll
            for (int j = 0; j < 4; ++j)
                acc[i][j] = __builtin_amdgcn_wmma_f32_16x16x32_f16(
                    false, a[i].v, false, b[j].v, (short)0, acc[i][j], false, false);
        __syncthreads();
    }
    // epilogue: bias + gelu -> H (f16)
    int rlo = (lane >> 4) * 8;
    int cl  = lane & 15;
#pragma unroll
    for (int i = 0; i < 2; ++i) {
        int mbase = m0 + wm * 32 + i * 16 + rlo;
#pragma unroll
        for (int j = 0; j < 4; ++j) {
            int n = n0 + wn * 64 + j * 16 + cl;
            float bias = b1[e * FF + n];
#pragma unroll
            for (int r = 0; r < 8; ++r) {
                float v = acc[i][j][r] + bias;
                H[((size_t)e * CAP + (size_t)(mbase + r)) * FF + n] = (_Float16)gelu_f(v);
            }
        }
    }
}

// ---------------------------------------------------------------- FFN2: out += g * (H*W2+b2)
__global__ __launch_bounds__(256) void ffn2_kernel(
        const _Float16* __restrict__ H, const float* __restrict__ w2,
        const float* __restrict__ b2, const int* __restrict__ disp,
        const int* __restrict__ tk_idx, const float* __restrict__ tk_gate,
        float* __restrict__ out) {
    __shared__ _Float16 As[128 * LDS_STRIDE];
    __shared__ _Float16 Bs[128 * LDS_STRIDE];
    __shared__ int   toks[128];
    __shared__ float gsh[128];

    int bid = blockIdx.x;
    int e   = bid / (MT1 * NT2);
    int rem = bid % (MT1 * NT2);
    int m0  = (rem / NT2) * 128;
    int n0  = (rem % NT2) * 128;
    int tid = threadIdx.x;
    int lane = tid & 31, wid = tid >> 5;
    int wm = wid & 3, wn = wid >> 2;

    if (tid < 128) {
        int tok = disp[e * CAP + m0 + tid];
        toks[tid] = tok;
        float g = 0.f;
        if (tok < TOKENS) {
            if (tk_idx[tok * 2] == e)          g = tk_gate[tok * 2];
            else if (tk_idx[tok * 2 + 1] == e) g = tk_gate[tok * 2 + 1];
        }
        gsh[tid] = g;
    }
    __syncthreads();

    v8f acc[2][4];
#pragma unroll
    for (int i = 0; i < 2; ++i)
#pragma unroll
        for (int j = 0; j < 4; ++j)
#pragma unroll
            for (int r = 0; r < 8; ++r) acc[i][j][r] = 0.f;

    const float* w2e = w2 + (size_t)e * FF * DIM;
    int arow  = tid >> 1;
    int acol  = (tid & 1) * 16;
    int brow  = tid & 31;
    int bcol0 = (tid >> 5) * 16;

    // per-thread fixed LDS destination for the async A-tile copy (f16, no convert)
    unsigned lds_dst = (unsigned)(size_t)(&As[arow * LDS_STRIDE + acol]);
    const _Float16* a_src_base =
        H + ((size_t)e * CAP + (size_t)(m0 + arow)) * FF + acol;

    for (int k0 = 0; k0 < FF; k0 += KB) {
        // stage A: H already f16 -> async copy straight into LDS (2 x b128)
        // GLOBAL_LOAD_ASYNC_TO_LDS_B128: per-lane LDS dest in VDST, global addr in
        // VADDR, IOFFSET applied to both sides; tracked by ASYNCcnt.
        {
            unsigned long long ga = (unsigned long long)(size_t)(a_src_base + k0);
            asm volatile(
                "global_load_async_to_lds_b128 %0, %1, off\n\t"
                "global_load_async_to_lds_b128 %0, %1, off offset:16"
                :: "v"(lds_dst), "v"(ga) : "memory");
        }
        // stage B: w2 tile fp32 -> f16, transpose to [n][k]
        {
            const float* gsrc = w2e + (size_t)(k0 + brow) * DIM + n0 + bcol0;
            if (k0 + KB < FF) __builtin_prefetch(gsrc + (size_t)KB * DIM, 0, 1);
            const float4* src = (const float4*)gsrc;
#pragma unroll
            for (int j = 0; j < 4; ++j) {
                float4 f = src[j];
                int n = bcol0 + j * 4;
                Bs[(n + 0) * LDS_STRIDE + brow] = (_Float16)f.x;
                Bs[(n + 1) * LDS_STRIDE + brow] = (_Float16)f.y;
                Bs[(n + 2) * LDS_STRIDE + brow] = (_Float16)f.z;
                Bs[(n + 3) * LDS_STRIDE + brow] = (_Float16)f.w;
            }
        }
        // drain this wave's async copies, then workgroup barrier
        asm volatile("s_wait_asynccnt 0x0" ::: "memory");
        __syncthreads();

        HFrag a[2], b[4];
        int sel = lane >> 4;
#pragma unroll
        for (int i = 0; i < 2; ++i) {
            const v8h* p = (const v8h*)&As[(wm * 32 + i * 16 + (lane & 15)) * LDS_STRIDE];
            a[i].h[0] = p[sel * 2];
            a[i].h[1] = p[sel * 2 + 1];
        }
#pragma unroll
        for (int j = 0; j < 4; ++j) {
            const v8h* p = (const v8h*)&Bs[(wn * 64 + j * 16 + (lane & 15)) * LDS_STRIDE + sel * 16];
            b[j].h[0] = p[0];
            b[j].h[1] = p[1];
        }
#pragma unroll
        for (int i = 0; i < 2; ++i)
#pragma unroll
            for (int j = 0; j < 4; ++j)
                acc[i][j] = __builtin_amdgcn_wmma_f32_16x16x32_f16(
                    false, a[i].v, false, b[j].v, (short)0, acc[i][j], false, false);
        __syncthreads();
    }
    // epilogue: bias, gate, atomic combine (dropped/padded rows skipped)
    int rlo = (lane >> 4) * 8;
    int cl  = lane & 15;
#pragma unroll
    for (int i = 0; i < 2; ++i) {
        int mloc = wm * 32 + i * 16 + rlo;
#pragma unroll
        for (int j = 0; j < 4; ++j) {
            int n = n0 + wn * 64 + j * 16 + cl;
            float bias = b2[e * DIM + n];
#pragma unroll
            for (int r = 0; r < 8; ++r) {
                int ml  = mloc + r;
                int tok = toks[ml];
                if (tok < TOKENS) {
                    float y = (acc[i][j][r] + bias) * gsh[ml];
                    atomicAdd(&out[(size_t)tok * DIM + n], y);
                }
            }
        }
    }
}

// ---------------------------------------------------------------- launch
extern "C" void kernel_launch(void* const* d_in, const int* in_sizes, int n_in,
                              void* d_out, int out_size, void* d_ws, size_t ws_size,
                              hipStream_t stream) {
    const float* x  = (const float*)d_in[0];
    const float* wr = (const float*)d_in[1];
    const float* w1 = (const float*)d_in[2];
    const float* b1 = (const float*)d_in[3];
    const float* w2 = (const float*)d_in[4];
    const float* b2 = (const float*)d_in[5];
    float* out = (float*)d_out;

    char* ws = (char*)d_ws;
    const size_t H_BYTES = (size_t)NEXP * CAP * FF * sizeof(_Float16); // ~168 MB
    _Float16* H      = (_Float16*)ws;
    int*      tk_idx = (int*)(ws + H_BYTES);
    float*    tk_gate= (float*)(ws + H_BYTES + (size_t)TOKENS * 2 * sizeof(int));
    int*      disp   = (int*)(ws + H_BYTES + (size_t)TOKENS * 2 * (sizeof(int) + sizeof(float)));

    int n4 = TOKENS * DIM / 4;
    zero_kernel<<<(n4 + 255) / 256, 256, 0, stream>>>((float4*)out, n4);
    router_kernel<<<TOKENS / 8, 256, 0, stream>>>(x, wr, tk_idx, tk_gate);
    dispatch_kernel<<<1, 256, 0, stream>>>(tk_idx, disp);
    ffn1_kernel<<<NEXP * MT1 * NT1, 256, 0, stream>>>(x, w1, b1, disp, H);
    ffn2_kernel<<<NEXP * MT1 * NT2, 256, 0, stream>>>(H, w2, b2, disp, tk_idx, tk_gate, out);
}
